// ResidualSNNBlock_49022756716719
// MI455X (gfx1250) — matrix operand order repaired and checked by reading, not compiled, not run on hardware
//
#include <hip/hip_runtime.h>
#include <hip/hip_bf16.h>
#include <cstddef>

// ---------------------------------------------------------------------------
// Problem constants (from reference): x [T=8,B=16,64,64,64] f32, ch=128.
// ---------------------------------------------------------------------------
#define T_STEPS 8
#define BATCH   16
#define TB      128           // T*B
#define CIN1    64
#define CO      128
#define HH      64
#define WW      64
#define HW      4096          // 64*64
#define NPIX    524288        // TB*HW  (BN normalization count)
#define PER_T   8388608       // B*CO*HW (stride between time steps in y buffers)

#define APAD    40            // padded LDS row (80B = 20 banks; gcd(20,64)=4 ->
                              // 16 consecutive rows hit 16 distinct banks)

typedef __attribute__((ext_vector_type(16))) __bf16 v16bf;
typedef __attribute__((ext_vector_type(8)))  __bf16 v8bf;
typedef __attribute__((ext_vector_type(2)))  __bf16 v2bf;
typedef __attribute__((ext_vector_type(8)))  float  v8f;

// ---------------------------------------------------------------------------
// Small utility kernels
// ---------------------------------------------------------------------------
__global__ __launch_bounds__(256) void zero_f32(float* p, int n) {
  int i = blockIdx.x * 256 + threadIdx.x;
  if (i < n) p[i] = 0.f;
}

__global__ __launch_bounds__(256) void cvt_f32_bf16(const float* __restrict__ in,
                                                    __bf16* __restrict__ out, int n) {
  int i = blockIdx.x * 256 + threadIdx.x;
  if (i < n) out[i] = (__bf16)in[i];
}

// w [CO][CIN][KH*KW] f32  ->  wp [KPOS][CO][CIN] bf16   (GEMM-friendly A tiles)
template <int CIN, int KPOS>
__global__ __launch_bounds__(256) void pack_w(const float* __restrict__ w,
                                              __bf16* __restrict__ wp) {
  int i = blockIdx.x * 256 + threadIdx.x;
  const int total = CO * CIN * KPOS;
  if (i >= total) return;
  int pos = i % KPOS;
  int ci  = (i / KPOS) % CIN;
  int co  = i / (KPOS * CIN);
  wp[((size_t)pos * CO + co) * CIN + ci] = (__bf16)w[((size_t)co * CIN + ci) * KPOS + pos];
}

// ---------------------------------------------------------------------------
// Implicit-GEMM conv + BN-stat reduction.
//   Workgroup: 256 threads (8 wave32), tile = 128 co x 64 px (one image row).
//   K order: (kh,kw) major, ci minor -> each K=32 chunk is one tap x 32 ch.
//   Double-buffered LDS (global->reg while WMMA runs), 1 barrier per chunk.
//   Padded LDS rows make A/B fragment reads bank-conflict free.
// ---------------------------------------------------------------------------
template <int CIN, int KPOS>
__global__ __launch_bounds__(256)
void conv_wmma(const __bf16* __restrict__ xin,   // [TB][CIN][HW] bf16
               const __bf16* __restrict__ wp,    // [KPOS][CO][CIN] bf16
               float* __restrict__ y,            // [TB][CO][HW] f32
               float* __restrict__ gSum,         // [CO]
               float* __restrict__ gSqr) {       // [CO]
  constexpr int CBLK   = CIN / 32;               // ci chunks per tap
  constexpr int NCHUNK = KPOS * CBLK;

  __shared__ __bf16 Ash[2][CO][APAD];
  __shared__ __bf16 Bsh[2][64][APAD];
  __shared__ float  sSum[CO];
  __shared__ float  sSqr[CO];

  const int tid  = threadIdx.x;
  const int wave = tid >> 5;
  const int lane = tid & 31;
  const int n    = blockIdx.x / HH;              // image (t*B+b)
  const int h    = blockIdx.x % HH;              // output row

  if (tid < CO) { sSum[tid] = 0.f; sSqr[tid] = 0.f; }

  v8f c[4] = {};                                 // 16co x 64px per wave

  // staging registers
  v8bf   ar0, ar1;                               // 16 weights / thread
  __bf16 br[8];                                  // 8 input px / thread

  const int acoL = tid >> 1;                     // A staging: co row
  const int accL = (tid & 1) << 4;               //            half-row offset
  const int bciL = tid & 31;                     // B staging: ci column
  const int bp0L = (tid >> 5) << 3;              //            px base

  auto loadRegs = [&](int chunk) {
    const int pos = chunk / CBLK;
    const int ci0 = (chunk % CBLK) << 5;
    const int kh  = (KPOS == 9) ? (pos / 3) : 1;
    const int kw  = (KPOS == 9) ? (pos % 3) : 1;
    const int h2  = h + kh - 1;
    {
      const __bf16* src = wp + (((size_t)pos * CO + acoL) * CIN + ci0 + accL);
      ar0 = *(const v8bf*)src;
      ar1 = *(const v8bf*)(src + 8);
    }
    {
      const __bf16* src = xin + ((size_t)(n * CIN + ci0 + bciL) * HW + h2 * WW);
      if (chunk + 1 < NCHUNK && (unsigned)h2 < (unsigned)HH)
        __builtin_prefetch(src + 32 * HW, 0, 1);   // global_prefetch_b8
#pragma unroll
      for (int i = 0; i < 8; ++i) {
        const int wv = bp0L + i + kw - 1;
        __bf16 v = (__bf16)0.f;
        if ((unsigned)h2 < (unsigned)HH && (unsigned)wv < (unsigned)WW) v = src[wv];
        br[i] = v;
      }
    }
  };

  auto storeLDS = [&](int buf) {
    *(v8bf*)&Ash[buf][acoL][accL]     = ar0;     // 80B rows keep 16B alignment
    *(v8bf*)&Ash[buf][acoL][accL + 8] = ar1;
#pragma unroll
    for (int i = 0; i < 8; ++i) Bsh[buf][bp0L + i][bciL] = br[i];
  };

  loadRegs(0);

  for (int chunk = 0; chunk < NCHUNK; ++chunk) {
    const int buf = chunk & 1;
    storeLDS(buf);
    __syncthreads();
    if (chunk + 1 < NCHUNK) loadRegs(chunk + 1);   // overlap with WMMA below

    // --- A fragment: 16-bit A 16x32 VGPR layout (ISA 7.12.2) ---
    const int row = (wave << 4) + (lane & 15);
    const int kb  = (lane < 16) ? 0 : 8;
    v16bf a;
#pragma unroll
    for (int j = 0; j < 8; ++j) {
      const int K0 = kb + ((j < 4) ? (2 * j) : (2 * j + 8));
      v2bf pr = *(const v2bf*)&Ash[buf][row][K0];
      a[2 * j]     = pr[0];
      a[2 * j + 1] = pr[1];
    }
    // --- B fragments (same K order as A fragment) + WMMA ---
    const int kb2 = (lane < 16) ? 0 : 8;
#pragma unroll
    for (int t = 0; t < 4; ++t) {
      const int nrow = (t << 4) + (lane & 15);
      v8bf blo = *(const v8bf*)&Bsh[buf][nrow][kb2];
      v8bf bhi = *(const v8bf*)&Bsh[buf][nrow][kb2 + 16];
      v16bf b;
#pragma unroll
      for (int e = 0; e < 8; ++e) { b[e] = blo[e]; b[e + 8] = bhi[e]; }
      c[t] = __builtin_amdgcn_wmma_f32_16x16x32_bf16(
          false, a, false, b, (short)0, c[t], false, false);
    }
  }

  // --- epilogue: store conv output + per-channel sum / sum^2 for BN ---
  const int pxb  = lane & 15;
  const int moff = (lane < 16) ? 0 : 8;
  float* yrow = y + (size_t)n * CO * HW + h * WW;
#pragma unroll
  for (int r = 0; r < 8; ++r) {
    const int co = (wave << 4) + r + moff;
    float s = 0.f, q = 0.f;
#pragma unroll
    for (int t = 0; t < 4; ++t) {
      const float v = c[t][r];
      yrow[(size_t)co * HW + (t << 4) + pxb] = v;
      s += v;
      q += v * v;
    }
    atomicAdd(&sSum[co], s);
    atomicAdd(&sSqr[co], q);
  }
  __syncthreads();
  if (tid < CO) {
    atomicAdd(&gSum[tid], sSum[tid]);
    atomicAdd(&gSqr[tid], sSqr[tid]);
  }
}

// ---------------------------------------------------------------------------
// BN finalize: scale = g * rsqrt(var+eps); shift = b - mean*scale
// ---------------------------------------------------------------------------
__global__ __launch_bounds__(128)
void bn_finalize(const float* __restrict__ sum, const float* __restrict__ sqr,
                 const float* __restrict__ gamma, const float* __restrict__ beta,
                 float* __restrict__ scale, float* __restrict__ shift) {
  const int ch = threadIdx.x;
  const float inv_n = 1.f / (float)NPIX;
  const float m   = sum[ch] * inv_n;
  const float var = sqr[ch] * inv_n - m * m;
  const float sc  = gamma[ch] * __frsqrt_rn(var + 1e-5f);
  scale[ch] = sc;
  shift[ch] = beta[ch] - m * sc;
}

// ---------------------------------------------------------------------------
// LIF1: BN1-normalize, scan over T (v=(v+x)/2, fire>=1, hard reset),
//       emit spikes as bf16 (exact: spikes are {0,1}).
// ---------------------------------------------------------------------------
__global__ __launch_bounds__(256)
void lif1(const float* __restrict__ y1, const float* __restrict__ scale,
          const float* __restrict__ shift, __bf16* __restrict__ s1) {
  const size_t i = (size_t)blockIdx.x * 256 + threadIdx.x;  // over B*CO*HW
  const int ch = (int)((i >> 12) & (CO - 1));
  const float sc = scale[ch], sh = shift[ch];
  float v = 0.f;
#pragma unroll
  for (int t = 0; t < T_STEPS; ++t) {
    const float x = y1[(size_t)t * PER_T + i] * sc + sh;
    v = 0.5f * (v + x);
    const float s = (v >= 1.f) ? 1.f : 0.f;
    v *= (1.f - s);
    s1[(size_t)t * PER_T + i] = (__bf16)s;
  }
}

// ---------------------------------------------------------------------------
// LIF2: BN2(conv2) + BNs(skip), add, scan over T, write final f32 spikes.
// ---------------------------------------------------------------------------
__global__ __launch_bounds__(256)
void lif2(const float* __restrict__ y2, const float* __restrict__ ysk,
          const float* __restrict__ sc2, const float* __restrict__ sh2,
          const float* __restrict__ scs, const float* __restrict__ shs,
          float* __restrict__ out) {
  const size_t i = (size_t)blockIdx.x * 256 + threadIdx.x;  // over B*CO*HW
  const int ch = (int)((i >> 12) & (CO - 1));
  const float a2 = sc2[ch], b2 = sh2[ch];
  const float as = scs[ch], bs = shs[ch];
  float v = 0.f;
#pragma unroll
  for (int t = 0; t < T_STEPS; ++t) {
    const size_t o = (size_t)t * PER_T + i;
    const float x = y2[o] * a2 + b2 + ysk[o] * as + bs;
    v = 0.5f * (v + x);
    const float s = (v >= 1.f) ? 1.f : 0.f;
    v *= (1.f - s);
    out[o] = s;
  }
}

// ---------------------------------------------------------------------------
// Launcher
// ---------------------------------------------------------------------------
static inline size_t align256(size_t x) { return (x + 255) & ~(size_t)255; }

extern "C" void kernel_launch(void* const* d_in, const int* in_sizes, int n_in,
                              void* d_out, int out_size, void* d_ws, size_t ws_size,
                              hipStream_t stream) {
  (void)in_sizes; (void)n_in; (void)out_size; (void)ws_size;
  const float* x   = (const float*)d_in[0];
  const float* w1  = (const float*)d_in[1];
  const float* g1  = (const float*)d_in[2];
  const float* b1  = (const float*)d_in[3];
  const float* w2  = (const float*)d_in[4];
  const float* g2  = (const float*)d_in[5];
  const float* b2  = (const float*)d_in[6];
  const float* wsk = (const float*)d_in[7];
  const float* gs  = (const float*)d_in[8];
  const float* bs  = (const float*)d_in[9];
  float* out = (float*)d_out;

  // workspace carve-up
  char* p = (char*)d_ws;
  size_t off = 0;
  const size_t nx = (size_t)TB * CIN1 * HW;   // 33.5M
  const size_t ny = (size_t)TB * CO * HW;     // 67.1M
  __bf16* xbf = (__bf16*)(p + off); off = align256(off + nx * 2);
  __bf16* wp1 = (__bf16*)(p + off); off = align256(off + (size_t)9 * CO * CIN1 * 2);
  __bf16* wp2 = (__bf16*)(p + off); off = align256(off + (size_t)9 * CO * CO * 2);
  __bf16* wps = (__bf16*)(p + off); off = align256(off + (size_t)CO * CIN1 * 2);
  float*  y1  = (float*)(p + off);  off = align256(off + ny * 4);
  __bf16* s1  = (__bf16*)(p + off); off = align256(off + ny * 2);
  float*  y2  = (float*)(p + off);  off = align256(off + ny * 4);
  float*  ysk = (float*)(p + off);  off = align256(off + ny * 4);
  float*  stats = (float*)(p + off);  // 12 * CO floats
  float *sum1 = stats,          *sqr1 = stats + CO,
        *sum2 = stats + 2 * CO, *sqr2 = stats + 3 * CO,
        *sums = stats + 4 * CO, *sqrs = stats + 5 * CO,
        *sc1  = stats + 6 * CO, *sh1  = stats + 7 * CO,
        *sc2  = stats + 8 * CO, *sh2  = stats + 9 * CO,
        *scs  = stats + 10 * CO, *shs = stats + 11 * CO;

  // 0) zero BN accumulators (deterministic each call)
  zero_f32<<<(12 * CO + 255) / 256, 256, 0, stream>>>(stats, 12 * CO);

  // 1) pack inputs/weights to bf16
  cvt_f32_bf16<<<(int)((nx + 255) / 256), 256, 0, stream>>>(x, xbf, (int)nx);
  pack_w<CIN1, 9><<<(9 * CO * CIN1 + 255) / 256, 256, 0, stream>>>(w1, wp1);
  pack_w<CO, 9><<<(9 * CO * CO + 255) / 256, 256, 0, stream>>>(w2, wp2);
  pack_w<CIN1, 1><<<(CO * CIN1 + 255) / 256, 256, 0, stream>>>(wsk, wps);

  const int convGrid = TB * HH;  // 8192 blocks, one (image,row) tile each

  // 2) conv1 (3x3, 64->128) + stats
  conv_wmma<CIN1, 9><<<convGrid, 256, 0, stream>>>(xbf, wp1, y1, sum1, sqr1);
  bn_finalize<<<1, 128, 0, stream>>>(sum1, sqr1, g1, b1, sc1, sh1);

  // 3) LIF1 -> bf16 spikes
  lif1<<<PER_T / 256, 256, 0, stream>>>(y1, sc1, sh1, s1);

  // 4) conv2 (3x3, 128->128) on spikes + stats; 1x1 skip conv + stats
  conv_wmma<CO, 9><<<convGrid, 256, 0, stream>>>(s1, wp2, y2, sum2, sqr2);
  conv_wmma<CIN1, 1><<<convGrid, 256, 0, stream>>>(xbf, wps, ysk, sums, sqrs);
  bn_finalize<<<1, 128, 0, stream>>>(sum2, sqr2, g2, b2, sc2, sh2);
  bn_finalize<<<1, 128, 0, stream>>>(sums, sqrs, gs, bs, scs, shs);

  // 5) LIF2 -> final f32 spikes
  lif2<<<PER_T / 256, 256, 0, stream>>>(y2, ysk, sc2, sh2, scs, shs, out);
}
